// MRTStep_50989851738406
// MI455X (gfx1250) — compile-verified, matrix operand order checked by reading, not run on hardware
//
#include <hip/hip_runtime.h>
#include <hip/hip_bf16.h>

// ---------------- constants (match reference) ----------------
#define VSZ  16000
#define EMB  512
#define HID  512
#define SLEN 24
#define TLEN 26
#define BB   2
#define NN   100
#define BN   200     // B*N
#define M0D  208     // BN padded to 16
#define M0E  16      // encoder rows padded to 16
#define H3   1536
#define STEPS 37     // MAX_LEN-1
#define TPB  256

// GEMM tiling
#define KC   128     // K chunk staged per TDM transfer
#define NBLK 64      // output columns per workgroup (4 N-tiles, one wave each M-tile)

typedef __attribute__((ext_vector_type(16))) __bf16 v16bf;
typedef __attribute__((ext_vector_type(8)))  float  v8f;
typedef __attribute__((ext_vector_type(4)))  unsigned v4u;
typedef __attribute__((ext_vector_type(8)))  unsigned v8u;

// ---------------- helpers ----------------
__device__ __forceinline__ float sigf(float x){ return 1.0f/(1.0f+__expf(-x)); }

__device__ __forceinline__ unsigned hash3(unsigned a, unsigned b, unsigned c){
  unsigned h = a*0x9E3779B1u;
  h ^= (b*0x85EBCA77u) + 0x9E3779B9u + (h<<6) + (h>>2);
  h ^= (c*0xC2B2AE3Du) + 0x27D4EB2Fu + (h<<5) + (h>>3);
  h ^= h >> 16; h *= 0x7FEB352Du; h ^= h >> 15; h *= 0x846CA68Bu; h ^= h >> 16;
  return h;
}

// Issue one TDM transfer: copy a [64 rows x KC elems] bf16 panel (row stride K elems)
// from global (Bt, row-major NxK) into LDS at byte offset ldsOff.
// D# per CDNA5 ISA ch.8: group0 = {count, lds_addr, global_addr, type=2},
// group1 = {data_size=2B, tensor dims (no OOB), tile = KC x 64, dim0 stride = K}.
__device__ __forceinline__ void tdm_load_panel(const __hip_bfloat16* gsrc,
                                               unsigned ldsOff, int K){
  unsigned long long ga = (unsigned long long)(size_t)gsrc;
  v4u g0;
  g0.x = 1u;                                           // count=1, user descriptor
  g0.y = ldsOff;                                       // lds_addr (bytes)
  g0.z = (unsigned)(ga & 0xFFFFFFFFu);                 // global_addr[31:0]
  g0.w = (unsigned)((ga >> 32) & 0x01FFFFFFu) | (2u << 30); // addr[56:32] | type=2
  v8u g1;
  g1[0] = (1u << 16);                                  // data_size=1 -> 2 bytes
  g1[1] = 0xFFFF0000u;                                 // tensor_dim0 lo16 (0x7FFFFFFF)
  g1[2] = 0xFFFF7FFFu;                                 // td0 hi16 | tensor_dim1 lo16
  g1[3] = ((unsigned)KC << 16) | 0x7FFFu;              // td1 hi16 | tile_dim0 = KC
  g1[4] = (unsigned)NBLK;                              // tile_dim1 = 64 rows (tile_dim2=0)
  g1[5] = (unsigned)K;                                 // tensor_dim0_stride lo32
  g1[6] = 0u;                                          // stride hi16 | dim1_stride lo16
  g1[7] = 0u;
  asm volatile("tensor_load_to_lds %0, %1" :: "s"(g0), "s"(g1) : "memory");
}

// ---------------- weight convert + transpose: W[K,N] f32 -> Wt[N,K] bf16 ----
__global__ void k_convT(const float* __restrict__ W, __hip_bfloat16* __restrict__ Wt,
                        int K, int N){
  int idx = blockIdx.x*blockDim.x + threadIdx.x;
  if (idx >= K*N) return;
  int k = idx / N, n = idx % N;
  Wt[(size_t)n*K + k] = __float2bfloat16(W[idx]);
}

__global__ void k_fill_f32(float* p, float v, int n){
  int i = blockIdx.x*blockDim.x + threadIdx.x;
  if (i < n) p[i] = v;
}
__global__ void k_fill_bf(__hip_bfloat16* p, int n){
  int i = blockIdx.x*blockDim.x + threadIdx.x;
  if (i < n) p[i] = __float2bfloat16(0.0f);
}

// ---------------- embedding gather (f32 table -> bf16 rows, zero pad rows) ---
__global__ void k_gather_emb(const int* __restrict__ tok, const float* __restrict__ emb,
                             __hip_bfloat16* __restrict__ x, int realRows, int M0, int E){
  int idx = blockIdx.x*blockDim.x + threadIdx.x;
  if (idx >= M0*E) return;
  int row = idx / E, e = idx - row*E;
  float v = 0.0f;
  if (row < realRows) v = emb[(size_t)tok[row]*E + e];
  x[idx] = __float2bfloat16(v);
}

// ---------------- WMMA GEMM: out = act(A[M0,K] * Bt[N,K]^T + bias) ----------
// Workgroup = 4 waves = 64x64 output block (wave w owns M-tile mb*4+w, all 4 N-tiles).
// B panel (64 cols x KC) staged to LDS by the Tensor Data Mover, double-buffered
// on TENSORcnt; A fragments stream from global (tiny, L2-hot).
// All four B fragments are loaded into distinct registers before the four WMMAs
// so the ds_loads clause together and the XDL pipe runs back-to-back.
__global__ void k_wmma_gemm(const __hip_bfloat16* __restrict__ A, int lda,
                            const __hip_bfloat16* __restrict__ Bt,
                            const float* __restrict__ bias,
                            float* __restrict__ outF, __hip_bfloat16* __restrict__ outBf,
                            int M0, int K, int Nn, int act){
  __shared__ __hip_bfloat16 ldsB[2][NBLK*KC];   // 2 x 16 KB

  const int lane   = threadIdx.x & 31;
  const int warp   = threadIdx.x >> 5;
  const int mTiles = M0 >> 4;
  const int mBlocks = (mTiles + 3) >> 2;
  const int mb = blockIdx.x % mBlocks;
  const int nb = blockIdx.x / mBlocks;

  int mt = mb*4 + warp;
  const bool active = (mt < mTiles);
  if (!active) mt = mTiles - 1;                 // clamp; keep wave in barriers

  const int nBase = nb * NBLK;
  const int rA    = mt*16 + (lane & 15);
  const int kHalf = (lane >> 4) * 8;            // ISA 16-bit fragment sub-block
  const __hip_bfloat16* aRow = A + (size_t)rA*lda + kHalf;

  const unsigned ldsBase = (unsigned)(size_t)(&ldsB[0][0]);  // flat addr[31:0] = LDS offset
  const unsigned bufBytes = NBLK*KC*2;
  const int nChunks = K / KC;

  if (warp == 0)
    tdm_load_panel(Bt + (size_t)nBase*K, ldsBase, K);        // chunk 0 -> buf 0

  v8f c[4] = {};
  for (int ci = 0; ci < nChunks; ++ci){
    if (warp == 0){
      if (ci + 1 < nChunks){
        tdm_load_panel(Bt + (size_t)nBase*K + (ci+1)*KC,
                       ldsBase + ((unsigned)(ci+1)&1u)*bufBytes, K);
        __builtin_amdgcn_s_wait_tensorcnt(1);   // chunk ci landed, ci+1 in flight
      } else {
        __builtin_amdgcn_s_wait_tensorcnt(0);
      }
    }
    __syncthreads();                            // publish chunk ci to all waves

    const __hip_bfloat16* bufp = &ldsB[ci & 1][0];
    for (int kk = 0; kk < KC; kk += 32){
      const int kg = ci*KC + kk;
      union { v16bf v; uint4 u[2]; } a, b[4];
      a.u[0] = *(const uint4*)(aRow + kg);
      a.u[1] = *(const uint4*)(aRow + kg + 16);
      if (kk + 32 < KC) __builtin_prefetch(aRow + kg + 32, 0, 1);
#pragma unroll
      for (int nt = 0; nt < 4; ++nt){           // gather all B frags first
        const __hip_bfloat16* bp = bufp + (size_t)(nt*16 + (lane & 15))*KC + kk + kHalf;
        b[nt].u[0] = *(const uint4*)bp;
        b[nt].u[1] = *(const uint4*)(bp + 16);
      }
#pragma unroll
      for (int nt = 0; nt < 4; ++nt)            // then 4 back-to-back WMMAs
        c[nt] = __builtin_amdgcn_wmma_f32_16x16x32_bf16(false, a.v, false, b[nt].v,
                                                        (short)0, c[nt], false, false);
    }
    __syncthreads();                            // all waves done before buffer reuse
  }

  if (!active) return;
  // C layout: VGPR j -> M = j + 8*(lane>=16), N = lane&15
  const int rowBase = mt*16 + (lane >> 4)*8;
#pragma unroll
  for (int nt = 0; nt < 4; ++nt){
    const int col = nBase + nt*16 + (lane & 15);
    const float bv = bias ? bias[col] : 0.0f;
#pragma unroll
    for (int j = 0; j < 8; ++j){
      float v = c[nt][j] + bv;
      if (act == 1) v = tanhf(v);
      size_t idx = (size_t)(rowBase + j)*Nn + col;
      if (outF)  outF[idx]  = v;
      if (outBf) outBf[idx] = __float2bfloat16(v);
    }
  }
}

// ---------------- GRU gate combine -----------------------------------------
__global__ void k_gru_gates(const float* __restrict__ gx, const float* __restrict__ gh,
                            const float* __restrict__ hin,
                            float* __restrict__ hout,
                            __hip_bfloat16* __restrict__ hbf, int ldbf,
                            float* __restrict__ ctxOut,
                            int realRows, int M0, int Hh){
  int idx = blockIdx.x*blockDim.x + threadIdx.x;
  if (idx >= M0*Hh) return;
  int row = idx / Hh, j = idx - row*Hh;
  float hn = 0.0f;
  if (row < realRows){
    const float* gxr = gx + (size_t)row*3*Hh;
    const float* ghr = gh + (size_t)row*3*Hh;
    float z = sigf(gxr[j]       + ghr[j]);
    float r = sigf(gxr[Hh+j]    + ghr[Hh+j]);
    float n = tanhf(gxr[2*Hh+j] + r*ghr[2*Hh+j]);
    float h = hin[(size_t)row*Hh + j];
    hn = (1.0f - z)*n + z*h;
  }
  hout[(size_t)row*Hh + j] = hn;
  hbf[(size_t)row*ldbf + j] = __float2bfloat16(hn);
  if (ctxOut && row < realRows) ctxOut[(size_t)row*Hh + j] = hn;
}

// ---------------- attention: scores->softmax->context, writes bf16 c into hc
__global__ void k_attn(const float* __restrict__ h, const float* __restrict__ ctx,
                       __hip_bfloat16* __restrict__ hc){
  __shared__ float sh[HID];
  __shared__ float sc[32];
  const int b = blockIdx.x, tid = threadIdx.x;
  const bool real = b < BN;
  const int borig = real ? (b / NN) : 0;

  for (int j = tid; j < HID; j += TPB) sh[j] = h[(size_t)b*HID + j];
  __syncthreads();

  if (tid < SLEN){
    const float* cr = ctx + ((size_t)tid*BB + borig)*HID;
    float d = 0.0f;
    for (int j = 0; j < HID; ++j) d += sh[j]*cr[j];
    sc[tid] = d;
  }
  __syncthreads();
  if (tid == 0){
    float mx = sc[0];
    for (int s = 1; s < SLEN; ++s) mx = fmaxf(mx, sc[s]);
    float sum = 0.0f;
    for (int s = 0; s < SLEN; ++s){ sc[s] = __expf(sc[s]-mx); sum += sc[s]; }
    float inv = 1.0f/sum;
    for (int s = 0; s < SLEN; ++s) sc[s] *= inv;
  }
  __syncthreads();
  for (int j = tid; j < HID; j += TPB){
    float c = 0.0f;
    for (int s = 0; s < SLEN; ++s) c += sc[s]*ctx[((size_t)s*BB + borig)*HID + j];
    hc[(size_t)b*(2*HID) + HID + j] = __float2bfloat16(real ? c : 0.0f);
  }
}

// ---------------- decoder state init: h = repeat(hT), tok=BOS ----------------
__global__ void k_init_dec(const float* __restrict__ hT, float* __restrict__ h,
                           __hip_bfloat16* __restrict__ hc, int* __restrict__ seq,
                           float* __restrict__ acc, float* __restrict__ dead){
  int idx = blockIdx.x*blockDim.x + threadIdx.x;
  if (idx >= M0D*HID) return;
  int row = idx / HID, j = idx - row*HID;
  float v = (row < BN) ? hT[(size_t)(row/NN)*HID + j] : 0.0f;
  h[idx] = v;
  hc[(size_t)row*(2*HID) + j] = __float2bfloat16(v);
  if (j == 0){ seq[row] = 2; acc[row] = 0.0f; dead[row] = 0.0f; }
}

// ---------------- sampling: logsoftmax + gumbel argmax + score accumulation --
__global__ void k_sample(const float* __restrict__ logits, int step,
                         int* __restrict__ seq, float* __restrict__ acc,
                         float* __restrict__ dead){
  __shared__ float rf[TPB];
  __shared__ int   ri[TPB];
  const int b = blockIdx.x, tid = threadIdx.x;
  const float* L = logits + (size_t)b*VSZ;

  float mx = -1e30f;
  for (int j = tid; j < VSZ; j += TPB) mx = fmaxf(mx, L[j]);
  rf[tid] = mx; __syncthreads();
  for (int s = TPB/2; s > 0; s >>= 1){ if (tid < s) rf[tid] = fmaxf(rf[tid], rf[tid+s]); __syncthreads(); }
  mx = rf[0]; __syncthreads();

  float sum = 0.0f;
  for (int j = tid; j < VSZ; j += TPB) sum += __expf(L[j]-mx);
  rf[tid] = sum; __syncthreads();
  for (int s = TPB/2; s > 0; s >>= 1){ if (tid < s) rf[tid] += rf[tid+s]; __syncthreads(); }
  const float lse = __logf(rf[0]); __syncthreads();

  float best = -1e30f; int bi = 0;
  for (int j = tid; j < VSZ; j += TPB){
    unsigned hsh = hash3((unsigned)step + 0x1234567u, (unsigned)b*40503u + 17u, (unsigned)j);
    float u = ((float)(hsh >> 8) + 0.5f) * (1.0f/16777216.0f);
    float g = -__logf(-__logf(u));
    float v = (L[j]-mx-lse) + g;
    if (v > best){ best = v; bi = j; }
  }
  rf[tid] = best; ri[tid] = bi; __syncthreads();
  for (int s = TPB/2; s > 0; s >>= 1){
    if (tid < s && rf[tid+s] > rf[tid]){ rf[tid] = rf[tid+s]; ri[tid] = ri[tid+s]; }
    __syncthreads();
  }
  if (tid == 0){
    int nxt = ri[0];
    float lp = L[nxt]-mx-lse;
    float d = dead[b];
    acc[b] += lp*(1.0f-d);
    dead[b] = fmaxf(d, (nxt==3) ? 1.0f : 0.0f);
    seq[(size_t)(step+1)*M0D + b] = nxt;
  }
}

// ---------------- final: BLEU, softmax over beams, 4 scalar outputs ----------
__global__ void k_final(const int* __restrict__ seq, const int* __restrict__ tgt,
                        const float* __restrict__ acc, float* __restrict__ out){
  __shared__ float bleu[BN];
  __shared__ float sred[TPB];
  __shared__ float swrd[TPB];
  const int tid = threadIdx.x;
  float totMatch = 0.0f, totWords = 0.0f;
  for (int row = tid; row < BN; row += TPB){
    int borig = row / NN;
    int m = 0, nw = 0;
    for (int t = 0; t < TLEN; ++t){
      int tg = tgt[t*BB + borig];
      if (tg != 0){ ++nw; if (seq[(size_t)t*M0D + row] == tg) ++m; }
    }
    bleu[row] = (float)m / (float)max(nw, 1);
    totMatch += (float)m;
    totWords += (float)nw;
  }
  sred[tid] = totMatch; swrd[tid] = totWords; __syncthreads();
  for (int s = TPB/2; s > 0; s >>= 1){
    if (tid < s){ sred[tid] += sred[tid+s]; swrd[tid] += swrd[tid+s]; }
    __syncthreads();
  }
  const float numCorrect = sred[0], numWords = swrd[0];
  __syncthreads();
  if (tid < BB){
    int base = tid*NN;
    float mx = -1e30f;
    for (int i = 0; i < NN; ++i) mx = fmaxf(mx, acc[base+i]*0.005f);
    float sum = 0.0f;
    for (int i = 0; i < NN; ++i) sum += __expf(acc[base+i]*0.005f - mx);
    float loss = 0.0f, sm = 0.0f;
    for (int i = 0; i < NN; ++i){
      float p = __expf(acc[base+i]*0.005f - mx)/sum;
      float bl = bleu[base+i];
      loss += bl*p; sm += bl*bl*p;
    }
    sred[tid] = loss; sred[64+tid] = sm;
  }
  __syncthreads();
  if (tid == 0){
    out[0] = -(sred[0]+sred[1]);
    out[1] = sred[64]+sred[65];
    out[2] = numWords;
    out[3] = numCorrect;
  }
}

// =====================================================================
extern "C" void kernel_launch(void* const* d_in, const int* in_sizes, int n_in,
                              void* d_out, int out_size, void* d_ws, size_t ws_size,
                              hipStream_t stream){
  (void)in_sizes; (void)n_in; (void)out_size; (void)ws_size;
  const int*   src     = (const int*)  d_in[0];
  const int*   tgt     = (const int*)  d_in[1];
  const float* emb_src = (const float*)d_in[3];
  const float* emb_tgt = (const float*)d_in[4];
  const float* Wex     = (const float*)d_in[5];
  const float* Weh     = (const float*)d_in[6];
  const float* be      = (const float*)d_in[7];
  const float* Wdx     = (const float*)d_in[8];
  const float* Wdh     = (const float*)d_in[9];
  const float* bd      = (const float*)d_in[10];
  const float* Wo      = (const float*)d_in[11];
  const float* bo      = (const float*)d_in[12];
  const float* Wg      = (const float*)d_in[13];
  const float* bg      = (const float*)d_in[14];

  char* ws = (char*)d_ws;
  size_t off = 0;
  auto alloc = [&](size_t bytes)->char*{
    char* p = ws + off; off = (off + bytes + 255) & ~(size_t)255; return p;
  };
  typedef __hip_bfloat16 bf;

  // bf16 transposed weights (N x K)
  bf* WexT = (bf*)alloc((size_t)H3*EMB*2);
  bf* WehT = (bf*)alloc((size_t)H3*HID*2);
  bf* WdxT = (bf*)alloc((size_t)H3*EMB*2);
  bf* WdhT = (bf*)alloc((size_t)H3*HID*2);
  bf* WoT  = (bf*)alloc((size_t)HID*(2*HID)*2);
  bf* WgT  = (bf*)alloc((size_t)VSZ*HID*2);
  // encoder state
  bf*    xe_bf  = (bf*)   alloc((size_t)M0E*EMB*2);
  float* he_f32 = (float*)alloc((size_t)M0E*HID*4);
  bf*    he_bf  = (bf*)   alloc((size_t)M0E*HID*2);
  float* gex    = (float*)alloc((size_t)M0E*H3*4);
  float* geh    = (float*)alloc((size_t)M0E*H3*4);
  float* ctx    = (float*)alloc((size_t)SLEN*BB*HID*4);
  // decoder state
  bf*    xd_bf  = (bf*)   alloc((size_t)M0D*EMB*2);
  float* h_f32  = (float*)alloc((size_t)M0D*HID*4);
  bf*    hc_bf  = (bf*)   alloc((size_t)M0D*(2*HID)*2);  // [h | c] bf16
  float* gdx    = (float*)alloc((size_t)M0D*H3*4);
  float* gdh    = (float*)alloc((size_t)M0D*H3*4);
  bf*    o_bf   = (bf*)   alloc((size_t)M0D*HID*2);
  float* logits = (float*)alloc((size_t)M0D*VSZ*4);
  int*   seq    = (int*)  alloc((size_t)(STEPS+1)*M0D*4);
  float* acc    = (float*)alloc((size_t)M0D*4);
  float* dead   = (float*)alloc((size_t)M0D*4);

  auto cdiv = [](int a, int b){ return (a + b - 1)/b; };
  auto gemm = [&](const bf* A, int lda, const bf* Bt, const float* bias,
                  float* outF, bf* outBf, int M0_, int K_, int N_, int act){
    int mBlocks = ((M0_/16) + 3) / 4;
    int nBlocks = N_ / NBLK;
    k_wmma_gemm<<<mBlocks*nBlocks, 128, 0, stream>>>(A, lda, Bt, bias, outF, outBf,
                                                     M0_, K_, N_, act);
  };

  // ---- weight conversion (each call; deterministic) ----
  k_convT<<<cdiv(EMB*H3,TPB),TPB,0,stream>>>(Wex, WexT, EMB, H3);
  k_convT<<<cdiv(HID*H3,TPB),TPB,0,stream>>>(Weh, WehT, HID, H3);
  k_convT<<<cdiv(EMB*H3,TPB),TPB,0,stream>>>(Wdx, WdxT, EMB, H3);
  k_convT<<<cdiv(HID*H3,TPB),TPB,0,stream>>>(Wdh, WdhT, HID, H3);
  k_convT<<<cdiv(2*HID*HID,TPB),TPB,0,stream>>>(Wo, WoT, 2*HID, HID);
  k_convT<<<cdiv(HID*VSZ,TPB),TPB,0,stream>>>(Wg, WgT, HID, VSZ);

  // ---- encoder ----
  k_fill_f32<<<cdiv(M0E*HID,TPB),TPB,0,stream>>>(he_f32, 0.0f, M0E*HID);
  k_fill_bf <<<cdiv(M0E*HID,TPB),TPB,0,stream>>>(he_bf, M0E*HID);
  for (int s = 0; s < SLEN; ++s){
    k_gather_emb<<<cdiv(M0E*EMB,TPB),TPB,0,stream>>>(src + s*BB, emb_src, xe_bf,
                                                     BB, M0E, EMB);
    gemm(xe_bf, EMB, WexT, be,      gex, nullptr, M0E, EMB, H3, 0);
    gemm(he_bf, HID, WehT, nullptr, geh, nullptr, M0E, HID, H3, 0);
    k_gru_gates<<<cdiv(M0E*HID,TPB),TPB,0,stream>>>(gex, geh, he_f32, he_f32,
                                                    he_bf, HID, ctx + (size_t)s*BB*HID,
                                                    BB, M0E, HID);
  }

  // ---- decoder init: hN = repeat(hT), tok0 = BOS ----
  k_init_dec<<<cdiv(M0D*HID,TPB),TPB,0,stream>>>(he_f32, h_f32, hc_bf, seq, acc, dead);

  // ---- decoder steps ----
  for (int t = 0; t < STEPS; ++t){
    k_gather_emb<<<cdiv(M0D*EMB,TPB),TPB,0,stream>>>(seq + (size_t)t*M0D, emb_tgt,
                                                     xd_bf, BN, M0D, EMB);
    gemm(xd_bf, EMB,    WdxT, bd,      gdx, nullptr, M0D, EMB, H3, 0);
    gemm(hc_bf, 2*HID,  WdhT, nullptr, gdh, nullptr, M0D, HID, H3, 0);
    k_gru_gates<<<cdiv(M0D*HID,TPB),TPB,0,stream>>>(gdx, gdh, h_f32, h_f32,
                                                    hc_bf, 2*HID, nullptr,
                                                    BN, M0D, HID);
    k_attn<<<M0D, TPB, 0, stream>>>(h_f32, ctx, hc_bf);
    gemm(hc_bf, 2*HID, WoT, bo, nullptr, o_bf, M0D, 2*HID, HID, 1);      // tanh
    gemm(o_bf,  HID,   WgT, bg, logits,  nullptr, M0D, HID, VSZ, 0);     // big GEMM
    k_sample<<<BN, TPB, 0, stream>>>(logits, t, seq, acc, dead);
  }

  // ---- loss / outputs ----
  k_final<<<1, TPB, 0, stream>>>(seq, tgt, acc, (float*)d_out);
}
